// TransformerDecoder_65506841199212
// MI455X (gfx1250) — matrix operand order repaired and checked by reading, not compiled, not run on hardware
//
#include <hip/hip_runtime.h>
#include <hip/hip_bf16.h>

// ---------------------------------------------------------------------------
// GPT-2-ish decoder forward for MI455X (gfx1250, wave32, WMMA).
// All GEMMs + attention einsums on v_wmma_f32_16x16x32_bf16 (f32 accum).
// Pure-copy LDS staging uses GLOBAL_LOAD_ASYNC_TO_LDS_B128 (ASYNCcnt path).
// ---------------------------------------------------------------------------

typedef __attribute__((ext_vector_type(16))) __bf16 v16bf;
typedef __attribute__((ext_vector_type(8)))  float  v8f;
typedef __attribute__((ext_vector_type(4)))  int    i32x4;

#define B_  2
#define T_  1024
#define D_  768
#define H_  12
#define HS_ 64
#define L_  6
#define V_  50257
#define MT  (B_ * T_)          // 2048 token rows

// GEMM tiling
#define BM  128
#define BN  128
#define BK  32
#define LDP 40                 // padded LDS pitch (elements) for 32-wide tiles

enum { MODE_QKV = 0, MODE_RES = 1, MODE_GELU = 2, MODE_F32 = 3 };

// --- async global->LDS copy path (guarded; falls back to uint4 staging) ----
#if defined(__has_builtin)
#if __has_builtin(__builtin_amdgcn_global_load_async_to_lds_b128) && \
    __has_builtin(__builtin_amdgcn_s_wait_asynccnt)
#define HAVE_ASYNC_LDS 1
#endif
#endif
#ifndef HAVE_ASYNC_LDS
#define HAVE_ASYNC_LDS 0
#endif

#if HAVE_ASYNC_LDS
// 16-byte async DMA: global -> LDS, tracked by ASYNCcnt.
// Builtin signature (from toolchain diagnostic): (v4i* global, v4i* lds, Ii, Ii)
template <int OFF>
__device__ inline void async_b128(const void* gsrc, void* ldst) {
    __builtin_amdgcn_global_load_async_to_lds_b128(
        (__attribute__((address_space(1))) i32x4*)(gsrc),
        (__attribute__((address_space(3))) i32x4*)(ldst),
        OFF, 0);
}
#endif

// cross-lane xor reduce step via ds_swizzle (group-of-32 xor mode)
template <int MASK>
__device__ inline float swz_xor(float v) {
    int i = __builtin_amdgcn_ds_swizzle(__builtin_bit_cast(int, v),
                                        0x1f | (MASK << 10));
    return __builtin_bit_cast(float, i);
}

// ---------------------------------------------------------------------------
// Embedding: x[m,d] = tok_embd[tokens[m], d] + pos_embd[m % T, d]
// ---------------------------------------------------------------------------
__global__ __launch_bounds__(256)
void embed_kernel(const int* __restrict__ tokens, const float* __restrict__ tok,
                  const float* __restrict__ pos, float* __restrict__ X) {
    int idx = blockIdx.x * 256 + threadIdx.x;     // grid covers MT*D exactly
    int m = idx / D_;
    int d = idx - m * D_;
    int t = m & (T_ - 1);
    int tk = tokens[m];
    X[idx] = tok[(size_t)tk * D_ + d] + pos[(size_t)t * D_ + d];
}

// ---------------------------------------------------------------------------
// LayerNorm over D=768, one block per token row, bf16 output.
// ---------------------------------------------------------------------------
__global__ __launch_bounds__(256)
void layernorm_bf16(const float* __restrict__ X, const float* __restrict__ g,
                    const float* __restrict__ bta, __bf16* __restrict__ Y) {
    __shared__ float red[256];
    const int m = blockIdx.x;
    const int tid = threadIdx.x;
    const float* xr = X + (size_t)m * D_;
    float v0 = xr[tid], v1 = xr[tid + 256], v2 = xr[tid + 512];
    red[tid] = v0 + v1 + v2;
    __syncthreads();
    for (int o = 128; o > 0; o >>= 1) {
        if (tid < o) red[tid] += red[tid + o];
        __syncthreads();
    }
    float mean = red[0] * (1.0f / D_);
    __syncthreads();
    float d0 = v0 - mean, d1 = v1 - mean, d2 = v2 - mean;
    red[tid] = d0 * d0 + d1 * d1 + d2 * d2;
    __syncthreads();
    for (int o = 128; o > 0; o >>= 1) {
        if (tid < o) red[tid] += red[tid + o];
        __syncthreads();
    }
    float rstd = rsqrtf(red[0] * (1.0f / D_) + 1e-5f);
    __bf16* yr = Y + (size_t)m * D_;
    float xv[3] = {v0, v1, v2};
#pragma unroll
    for (int i = 0; i < 3; i++) {
        int d = tid + i * 256;
        float y = (xv[i] - mean) * rstd * g[d];
        if (bta) y += bta[d];
        yr[d] = (__bf16)y;
    }
}

// ---------------------------------------------------------------------------
// WMMA GEMM: C[m,n] = sum_k A[m,k] * W[n,k]   (A bf16 [MT,K], W fp32 [N,K])
// 256 threads = 8 waves; block tile 128x128; wave tile 64(M) x 32(N);
// K staged 32 at a time through padded LDS. A tile staged by async-to-LDS
// DMA path (ASYNCcnt); W tile converted fp32->bf16 on the load path.
// ---------------------------------------------------------------------------
template <int MODE>
__global__ __launch_bounds__(256)
void gemm_wmma(const __bf16* __restrict__ A, const float* __restrict__ W,
               const float* __restrict__ bias, float* __restrict__ outF,
               __bf16* __restrict__ outB, int N, int K) {
    __shared__ __bf16 As[BM * LDP];
    __shared__ __bf16 Ws[BN * LDP];

    const int tid   = threadIdx.x;
    const int lane  = tid & 31;
    const int wave  = tid >> 5;
    const int waveM = wave >> 2;          // 0..1
    const int waveN = wave & 3;           // 0..3
    const int l15   = lane & 15;
    const int khalf = lane >> 4;
    const int n0 = blockIdx.x * BN;
    const int m0 = blockIdx.y * BM;

    v8f acc[4][2];
#pragma unroll
    for (int i = 0; i < 4; i++)
#pragma unroll
        for (int j = 0; j < 2; j++)
#pragma unroll
            for (int e = 0; e < 8; e++) acc[i][j][e] = 0.0f;

    const int arow = tid >> 1;            // 0..127
    const int acol = (tid & 1) * 16;      // 0 or 16

    for (int k0 = 0; k0 < K; k0 += BK) {
        __syncthreads();
        // stage A tile: 128 x 32 bf16 (each thread copies 16 elems = 32B)
        {
            const __bf16* gsrc = A + (size_t)(m0 + arow) * K + k0 + acol;
            __bf16* ldst = As + arow * LDP + acol;
#if HAVE_ASYNC_LDS
            async_b128<0>(gsrc, ldst);
            async_b128<16>(gsrc, ldst);
#else
            const uint4* src = (const uint4*)gsrc;
            uint4 d0 = src[0];
            uint4 d1 = src[1];
            uint4* dst = (uint4*)ldst;
            dst[0] = d0;
            dst[1] = d1;
#endif
        }
        // stage W tile: 128 x 32, fp32 -> bf16 on the fly, N-edge zero fill
        {
            int nr = n0 + arow;
            v16bf wv;
            if (nr < N) {
                const float4* wsrc = (const float4*)(W + (size_t)nr * K + k0 + acol);
#pragma unroll
                for (int c = 0; c < 4; c++) {
                    float4 f = wsrc[c];
                    wv[c * 4 + 0] = (__bf16)f.x;
                    wv[c * 4 + 1] = (__bf16)f.y;
                    wv[c * 4 + 2] = (__bf16)f.z;
                    wv[c * 4 + 3] = (__bf16)f.w;
                }
            } else {
#pragma unroll
                for (int e = 0; e < 16; e++) wv[e] = (__bf16)0.0f;
            }
            uint4* wd = (uint4*)(Ws + arow * LDP + acol);
            wd[0] = ((uint4*)&wv)[0];
            wd[1] = ((uint4*)&wv)[1];
        }
        // prefetch next K tile into cache (global_prefetch_b8)
        if (k0 + BK < K) {
            __builtin_prefetch(A + (size_t)(m0 + arow) * K + k0 + BK + acol, 0, 1);
            if (n0 + arow < N)
                __builtin_prefetch(W + (size_t)(n0 + arow) * K + k0 + BK + acol, 0, 1);
        }
#if HAVE_ASYNC_LDS
        __builtin_amdgcn_s_wait_asynccnt(0);   // A-tile DMA landed in LDS
#endif
        __syncthreads();

        // fragments: A 16x32 (two contiguous 8-elem K chunks per lane),
        //            B 32x16 (col = lane&15, same two 8-elem K chunks as A)
        v16bf aF[4], bF[2];
#pragma unroll
        for (int mi = 0; mi < 4; mi++) {
            const __bf16* p = As + (waveM * 64 + mi * 16 + l15) * LDP;
            ((uint4*)&aF[mi])[0] = *(const uint4*)(p + khalf * 8);
            ((uint4*)&aF[mi])[1] = *(const uint4*)(p + 16 + khalf * 8);
        }
#pragma unroll
        for (int ni = 0; ni < 2; ni++) {
            const __bf16* p = Ws + (waveN * 32 + ni * 16 + l15) * LDP;
            ((uint4*)&bF[ni])[0] = *(const uint4*)(p + khalf * 8);
            ((uint4*)&bF[ni])[1] = *(const uint4*)(p + 16 + khalf * 8);
        }
#pragma unroll
        for (int mi = 0; mi < 4; mi++)
#pragma unroll
            for (int ni = 0; ni < 2; ni++)
                acc[mi][ni] = __builtin_amdgcn_wmma_f32_16x16x32_bf16(
                    false, aF[mi], false, bF[ni], (short)0, acc[mi][ni], false, false);
    }

    // epilogue — C layout: lane holds (m = base + r + 8*(lane>>4), n = base + lane&15)
#pragma unroll
    for (int mi = 0; mi < 4; mi++) {
#pragma unroll
        for (int ni = 0; ni < 2; ni++) {
#pragma unroll
            for (int r = 0; r < 8; r++) {
                int m = m0 + waveM * 64 + mi * 16 + r + 8 * khalf;
                int n = n0 + waveN * 32 + ni * 16 + l15;
                if (n >= N) continue;
                float val = acc[mi][ni][r];
                if constexpr (MODE == MODE_QKV) {
                    // write bf16 into [B,H,T,HS] layout for attention
                    int b = m >> 10, t = m & (T_ - 1);
                    int hh = n >> 6, dd = n & 63;
                    outB[(((size_t)(b * H_ + hh)) * T_ + t) * HS_ + dd] = (__bf16)val;
                } else if constexpr (MODE == MODE_RES) {
                    size_t idx = (size_t)m * N + n;
                    outF[idx] = outF[idx] + bias[n] + val;   // residual in-place
                } else if constexpr (MODE == MODE_GELU) {
                    float gsum = val + bias[n];
                    float y = 0.5f * gsum * (1.0f + erff(gsum * 0.70710678118654752f));
                    outB[(size_t)m * N + n] = (__bf16)y;
                } else {
                    outF[(size_t)m * N + n] = val;
                }
            }
        }
    }
}

// ---------------------------------------------------------------------------
// Flash attention: one wave (32 threads) per (b, h, 16-query tile).
// QK^T and P*V both on WMMA; online softmax; P routed LDS C-layout->A-layout.
// V tile copy issued asynchronously BEFORE the softmax math; waited just
// before the P*V fragment gather (copy/compute overlap inside one wave).
// ---------------------------------------------------------------------------
__global__ __launch_bounds__(32)
void attn_wmma(const __bf16* __restrict__ Q, const __bf16* __restrict__ Kk,
               const __bf16* __restrict__ Vv, __bf16* __restrict__ O) {
    __shared__ __bf16 lds_p[16 * 40];
    __shared__ __bf16 lds_v[32 * 72];
    const int lane = threadIdx.x;
    const int l15 = lane & 15, khalf = lane >> 4;
    const int qt = blockIdx.x, h = blockIdx.y, b = blockIdx.z;
    const int q0 = qt * 16;
    const size_t headoff = (size_t)(b * H_ + h) * T_ * HS_;
    const __bf16* qp = Q + headoff;
    const __bf16* kp = Kk + headoff;
    const __bf16* vp = Vv + headoff;

    // Q fragments for d-chunks {0..31},{32..63}
    v16bf qF[2];
#pragma unroll
    for (int c = 0; c < 2; c++) {
        const __bf16* p = qp + (size_t)(q0 + l15) * HS_ + c * 32;
        ((uint4*)&qF[c])[0] = *(const uint4*)(p + khalf * 8);
        ((uint4*)&qF[c])[1] = *(const uint4*)(p + 16 + khalf * 8);
    }

    v8f accO[4];
#pragma unroll
    for (int d = 0; d < 4; d++)
#pragma unroll
        for (int e = 0; e < 8; e++) accO[d][e] = 0.0f;
    float mrow[8], lrow[8];
#pragma unroll
    for (int r = 0; r < 8; r++) { mrow[r] = -1e30f; lrow[r] = 0.0f; }

    const float scale = 0.125f;   // HS^-0.5

    for (int kb = 0; kb < q0 + 16; kb += 32) {
        // kick off V tile 32x64 copy into LDS (async DMA path if available)
        {
            const __bf16* gsrc = vp + (size_t)(kb + lane) * HS_;
            __bf16* ldst = lds_v + lane * 72;
#if HAVE_ASYNC_LDS
            async_b128<0>(gsrc, ldst);
            async_b128<16>(gsrc, ldst);
            async_b128<32>(gsrc, ldst);
            async_b128<48>(gsrc, ldst);
            async_b128<64>(gsrc, ldst);
            async_b128<80>(gsrc, ldst);
            async_b128<96>(gsrc, ldst);
            async_b128<112>(gsrc, ldst);
#else
            const uint4* src = (const uint4*)gsrc;
            uint4* dst = (uint4*)ldst;
#pragma unroll
            for (int c = 0; c < 8; c++) dst[c] = src[c];
#endif
        }
        // S = Q K^T for this 32-key block (two 16x16 tiles)
        v8f s[2];
#pragma unroll
        for (int jt = 0; jt < 2; jt++)
#pragma unroll
            for (int e = 0; e < 8; e++) s[jt][e] = 0.0f;
#pragma unroll
        for (int jt = 0; jt < 2; jt++) {
#pragma unroll
            for (int c = 0; c < 2; c++) {
                v16bf kF;   // B = K^T: col j = lane&15, d chunks (lane>>4)*8.. and 16+(lane>>4)*8..
                const __bf16* p = kp + (size_t)(kb + jt * 16 + l15) * HS_ + c * 32;
                ((uint4*)&kF)[0] = *(const uint4*)(p + khalf * 8);
                ((uint4*)&kF)[1] = *(const uint4*)(p + 16 + khalf * 8);
                s[jt] = __builtin_amdgcn_wmma_f32_16x16x32_bf16(
                    false, qF[c], false, kF, (short)0, s[jt], false, false);
            }
        }
        // causal mask + online softmax (row = q0 + r + 8*khalf)
#pragma unroll
        for (int r = 0; r < 8; r++) {
            const int row = q0 + r + 8 * khalf;
            float s0 = s[0][r] * scale;
            float s1 = s[1][r] * scale;
            if (kb + l15 > row)      s0 = -1e30f;
            if (kb + 16 + l15 > row) s1 = -1e30f;
            float mx = fmaxf(s0, s1);
            mx = fmaxf(mx, swz_xor<1>(mx));
            mx = fmaxf(mx, swz_xor<2>(mx));
            mx = fmaxf(mx, swz_xor<4>(mx));
            mx = fmaxf(mx, swz_xor<8>(mx));
            float mnew = fmaxf(mrow[r], mx);
            float alpha = __expf(mrow[r] - mnew);
            mrow[r] = mnew;
            float p0 = __expf(s0 - mnew);
            float p1 = __expf(s1 - mnew);
            float ps = p0 + p1;
            ps += swz_xor<1>(ps);
            ps += swz_xor<2>(ps);
            ps += swz_xor<4>(ps);
            ps += swz_xor<8>(ps);
            lrow[r] = lrow[r] * alpha + ps;
#pragma unroll
            for (int d = 0; d < 4; d++) accO[d][r] = accO[d][r] * alpha;
            // P tile (16q x 32k) to LDS in row-major for A-layout reload
            lds_p[(r + 8 * khalf) * 40 + l15]      = (__bf16)p0;
            lds_p[(r + 8 * khalf) * 40 + 16 + l15] = (__bf16)p1;
        }
        // P as A-fragment (16x32): same single wave, DScnt ordering by compiler
        v16bf pF;
        {
            const __bf16* p = lds_p + l15 * 40;
            ((uint4*)&pF)[0] = *(const uint4*)(p + khalf * 8);
            ((uint4*)&pF)[1] = *(const uint4*)(p + 16 + khalf * 8);
        }
#if HAVE_ASYNC_LDS
        __builtin_amdgcn_s_wait_asynccnt(0);   // V tile landed in LDS
#endif
        // O += P @ V  (B = V: col d = dslice*16 + lane&15, key = table order matching P)
#pragma unroll
        for (int d = 0; d < 4; d++) {
            v16bf vF;
#pragma unroll
            for (int e = 0; e < 16; e++)
                vF[e] = lds_v[((e < 8 ? 0 : 8) + khalf * 8 + e) * 72 + d * 16 + l15];
            accO[d] = __builtin_amdgcn_wmma_f32_16x16x32_bf16(
                false, pF, false, vF, (short)0, accO[d], false, false);
        }
    }
    // write O back token-major bf16 [MT, D]: concat heads
#pragma unroll
    for (int d = 0; d < 4; d++) {
#pragma unroll
        for (int r = 0; r < 8; r++) {
            int t = q0 + r + 8 * khalf;
            float val = accO[d][r] / lrow[r];
            O[((size_t)(b * T_ + t)) * D_ + h * HS_ + d * 16 + l15] = (__bf16)val;
        }
    }
}

// ---------------------------------------------------------------------------
// Host-side orchestration (graph-capture safe: only kernel launches on stream)
// ---------------------------------------------------------------------------
extern "C" void kernel_launch(void* const* d_in, const int* in_sizes, int n_in,
                              void* d_out, int out_size, void* d_ws, size_t ws_size,
                              hipStream_t stream) {
    const int*   tokens = (const int*)d_in[0];
    const float* tok_e  = (const float*)d_in[1];
    const float* pos_e  = (const float*)d_in[2];
    const float* Wq     = (const float*)d_in[3];
    const float* Wk     = (const float*)d_in[4];
    const float* Wv     = (const float*)d_in[5];
    const float* Wo     = (const float*)d_in[6];
    const float* bo     = (const float*)d_in[7];
    const float* g1     = (const float*)d_in[8];
    const float* W1     = (const float*)d_in[9];
    const float* b1     = (const float*)d_in[10];
    const float* W2     = (const float*)d_in[11];
    const float* b2     = (const float*)d_in[12];
    const float* g2     = (const float*)d_in[13];
    const float* ln_g   = (const float*)d_in[14];
    const float* ln_b   = (const float*)d_in[15];
    const float* Wout   = (const float*)d_in[16];
    float* out = (float*)d_out;

    // workspace carve-up (~35 MB)
    char* ws = (char*)d_ws;
    float*  X   = (float*)ws;   ws += (size_t)MT * D_ * sizeof(float);
    __bf16* Hb  = (__bf16*)ws;  ws += (size_t)MT * D_ * 2;
    __bf16* Qb  = (__bf16*)ws;  ws += (size_t)MT * D_ * 2;
    __bf16* Kb  = (__bf16*)ws;  ws += (size_t)MT * D_ * 2;
    __bf16* Vb  = (__bf16*)ws;  ws += (size_t)MT * D_ * 2;
    __bf16* Ob  = (__bf16*)ws;  ws += (size_t)MT * D_ * 2;
    __bf16* Mid = (__bf16*)ws;  ws += (size_t)MT * 4 * D_ * 2;

    dim3 blk(256);
    embed_kernel<<<MT * D_ / 256, blk, 0, stream>>>(tokens, tok_e, pos_e, X);

    dim3 gD((D_ + BN - 1) / BN, MT / BM);        // 6 x 16
    dim3 gF((4 * D_ + BN - 1) / BN, MT / BM);    // 24 x 16
    for (int l = 0; l < L_; ++l) {
        layernorm_bf16<<<MT, blk, 0, stream>>>(X, g1 + l * D_, nullptr, Hb);
        gemm_wmma<MODE_QKV><<<gD, blk, 0, stream>>>(Hb, Wq + (size_t)l * D_ * D_, nullptr, nullptr, Qb, D_, D_);
        gemm_wmma<MODE_QKV><<<gD, blk, 0, stream>>>(Hb, Wk + (size_t)l * D_ * D_, nullptr, nullptr, Kb, D_, D_);
        gemm_wmma<MODE_QKV><<<gD, blk, 0, stream>>>(Hb, Wv + (size_t)l * D_ * D_, nullptr, nullptr, Vb, D_, D_);
        attn_wmma<<<dim3(T_ / 16, H_, B_), dim3(32), 0, stream>>>(Qb, Kb, Vb, Ob);
        gemm_wmma<MODE_RES><<<gD, blk, 0, stream>>>(Ob, Wo + (size_t)l * D_ * D_, bo + l * D_, X, nullptr, D_, D_);
        layernorm_bf16<<<MT, blk, 0, stream>>>(X, g2 + l * D_, nullptr, Hb);
        gemm_wmma<MODE_GELU><<<gF, blk, 0, stream>>>(Hb, W1 + (size_t)l * 4 * D_ * D_, b1 + l * 4 * D_, nullptr, Mid, 4 * D_, D_);
        gemm_wmma<MODE_RES><<<gD, blk, 0, stream>>>(Mid, W2 + (size_t)l * D_ * 4 * D_, b2 + l * D_, X, nullptr, D_, 4 * D_);
    }
    layernorm_bf16<<<MT, blk, 0, stream>>>(X, ln_g, ln_b, Hb);
    dim3 gV((V_ + BN - 1) / BN, MT / BM);        // 393 x 16
    gemm_wmma<MODE_F32><<<gV, blk, 0, stream>>>(Hb, Wout, nullptr, out, nullptr, V_, D_);
}
